// MNIST_Monarch_MLP_30709016166958
// MI455X (gfx1250) — compile-verified, matrix operand order checked by reading, not run on hardware
//
#include <hip/hip_runtime.h>
#include <hip/hip_bf16.h>
#include <math.h>

// ---------------------------------------------------------------------------
// MI455X fused Monarch-MLP (4 layers) using V_WMMA_F32_16X16X32_F16.
// - memory floor: 205MB x read @ 23.3TB/s ~= 9us; f16 WMMA (f32 accum) keeps
//   matrix compute under that floor (f32 WMMA would not).
// - prep kernel packs all weight blocks into per-lane WMMA B-fragment order
//   in d_ws (zero-padded K/N), so the main kernel streams x once, keeps all
//   activations in LDS (f16), and never touches HBM for intermediates.
// - GELU uses a branch-free rational erf (|err|<=1.5e-7) built on
//   v_rcp_f32/v_exp_f32 instead of the branchy libm erff.
// ---------------------------------------------------------------------------

typedef _Float16 v16h __attribute__((ext_vector_type(16)));
typedef _Float16 v8h  __attribute__((ext_vector_type(8)));
typedef float    v8f  __attribute__((ext_vector_type(8)));

// K index within a 32-wide WMMA step for half h (0..15) of a lane.
// 16-bit operand layout (CDNA5 ISA): lanes 0-15 hold K {0..7,16..23},
// lanes 16-31 hold K {8..15,24..31}; 2 halves per VGPR in K order.
__device__ __host__ __forceinline__ int koff16(int h, int laneHi) {
  return (h < 8) ? (h + (laneHi ? 8 : 0)) : ((h - 8) + (laneHi ? 24 : 16));
}

// ---------------------------------------------------------------------------
// Prep: repack w[(blk*ND + n)*KD + kk] (f32) into f16 B-fragments.
// Fragment = 32 lanes x 16 halves (1024B), lane-major, zero padded for
// n >= ND or kk >= KD.  Fragment order: [blk][nt][ks].
// ---------------------------------------------------------------------------
__global__ void monarch_prep_frags(const float* __restrict__ w,
                                   _Float16* __restrict__ dst,
                                   int nblk, int ND, int KD, int NT, int KS) {
  int id = blockIdx.x * blockDim.x + threadIdx.x;
  int total = nblk * NT * KS * 512;
  if (id >= total) return;
  int h    = id & 15;
  int lane = (id >> 4) & 31;
  int fi   = id >> 9;                 // local fragment index
  int ks   = fi % KS;
  int nt   = (fi / KS) % NT;
  int blk  = fi / (KS * NT);
  int nn   = nt * 16 + (lane & 15);
  int kk   = ks * 32 + koff16(h, lane >= 16);
  float v = 0.f;
  if (nn < ND && kk < KD) v = w[(blk * ND + nn) * KD + kk];
  dst[id] = (_Float16)v;
}

// Load a 16-half A-fragment from LDS: two aligned b128 runs (base, base+16).
__device__ __forceinline__ v16h load_a16(const _Float16* p) {
  v8h lo = *(const v8h*)(p);
  v8h hi = *(const v8h*)(p + 16);
  return __builtin_shufflevector(lo, hi, 0,1,2,3,4,5,6,7,8,9,10,11,12,13,14,15);
}

// Exact-erf GELU, branch-free (A&S 7.1.26, |abs err| <= 1.5e-7).
__device__ __forceinline__ float gelu_exact(float v) {
  float ax = __builtin_fabsf(v) * 0.70710678118654752f;   // |v|/sqrt(2)
  float t  = __builtin_amdgcn_rcpf(__builtin_fmaf(0.3275911f, ax, 1.0f));
  float p  = __builtin_fmaf(1.061405429f, t, -1.453152027f);
  p = __builtin_fmaf(p, t, 1.421413741f);
  p = __builtin_fmaf(p, t, -0.284496736f);
  p = __builtin_fmaf(p, t, 0.254829592f);
  p = p * t;
  float e = __expf(-ax * ax);
  float half_erfc = 0.5f * p * e;                          // 0.5*erfc(ax)
  float cdf = (v >= 0.f) ? (1.f - half_erfc) : half_erfc;  // Phi(v)
  return v * cdf;
}

// ---------------------------------------------------------------------------
// One Monarch layer on a 16-row batch tile held in LDS.
//   mm1: out1[b,k,q] = sum_p A[b, k, p] * w1[k,q,p]   (WMMA, K padded)
//   permute j=k*QD+q -> (l=j%15, r=j/15), scatter into bufT1 [row][l][TSTR]
//   mm2: out2[b,l,s] = sum_r t1[b,l,r] * w2[l,s,r]
//   final permute f = s*15+l; f<OUTF: gelu(v+bias) else 0 -> next layer slots
// ---------------------------------------------------------------------------
template<int SBLK_IN, int KS1, int QD, int NT1,
         int TSTR, int SD, int NT2,
         int FB1, int FB2,
         int OUTF, int SBLK_OUT, int PD_OUT, bool LAST>
__device__ __forceinline__ void run_layer(int lane,
                                          _Float16* bufA, _Float16* bufT1,
                                          float* logits,
                                          const _Float16* __restrict__ wsh,
                                          const float* __restrict__ bias) {
  const int n     = lane & 15;          // C column / A row
  const int hoff  = (lane >> 4) * 8;    // K sub-offset for A fragments
  const int mbase = (lane >> 4) * 8;    // C row group

  // ---- mm1 ----
  for (int k = 0; k < 15; ++k) {
    v16h a[KS1];
#pragma unroll
    for (int ks = 0; ks < KS1; ++ks)
      a[ks] = load_a16(bufA + (n * 15 + k) * SBLK_IN + ks * 32 + hoff);
#pragma unroll
    for (int nt = 0; nt < NT1; ++nt) {
      v8f c = {};
#pragma unroll
      for (int ks = 0; ks < KS1; ++ks) {
        v16h b = *(const v16h*)(wsh +
                  (size_t)(FB1 + (k * NT1 + nt) * KS1 + ks) * 512 + lane * 16);
        c = __builtin_amdgcn_wmma_f32_16x16x32_f16(false, a[ks], false, b,
                                                   (short)0, c, false, false);
      }
      const int q = nt * 16 + n;
      if (q < QD) {
        const int j = k * QD + q;
        const int l = j % 15, r = j / 15;
#pragma unroll
        for (int v = 0; v < 8; ++v)
          bufT1[((mbase + v) * 15 + l) * TSTR + r] = (_Float16)c[v];
      }
    }
  }
  __syncthreads();

  // ---- mm2 ----
  for (int l = 0; l < 15; ++l) {
    v16h a = load_a16(bufT1 + (n * 15 + l) * TSTR + hoff);
#pragma unroll
    for (int nt = 0; nt < NT2; ++nt) {
      v8f c = {};
      v16h b = *(const v16h*)(wsh + (size_t)(FB2 + l * NT2 + nt) * 512 + lane * 16);
      c = __builtin_amdgcn_wmma_f32_16x16x32_f16(false, a, false, b,
                                                 (short)0, c, false, false);
      const int s = nt * 16 + n;
      if (s < SD) {
        const int f = s * 15 + l;
#pragma unroll
        for (int v = 0; v < 8; ++v) {
          const int row = mbase + v;
          if constexpr (LAST) {
            if (f < OUTF) logits[row * 16 + f] = c[v] + bias[f];
          } else {
            float g = 0.f;
            if (f < OUTF) g = gelu_exact(c[v] + bias[f]);
            const int k2 = f / PD_OUT, p2 = f % PD_OUT;
            bufA[(row * 15 + k2) * SBLK_OUT + p2] = (_Float16)g;
          }
        }
      }
    }
  }
  __syncthreads();
}

// ---------------------------------------------------------------------------
// Main fused kernel: one wave (32 threads) per 16-row batch tile.
// ---------------------------------------------------------------------------
__global__ __launch_bounds__(32)
void monarch_mlp_main(const float* __restrict__ x,
                      const _Float16* __restrict__ wsh,
                      const float* __restrict__ b1, const float* __restrict__ b2,
                      const float* __restrict__ b3, const float* __restrict__ b4,
                      float* __restrict__ out, int nB) {
  __shared__ __align__(32) _Float16 bufA[16 * 15 * 64];   // 30720B, act blocks
  __shared__ __align__(32) _Float16 bufT1[16 * 15 * 32];  // 15360B, t1 blocks
  __shared__ float logits[16 * 16];

  const int lane  = threadIdx.x;
  const int tile  = blockIdx.x;
  const int grow0 = tile * 16;

  // Zero bufT1 once (pad slots feed WMMA against zero-B; must be finite).
  {
    v8h z = {};
    for (int i = lane * 8; i < 16 * 15 * 32; i += 32 * 8)
      *(v8h*)(bufT1 + i) = z;
  }

  // Stage x tile into bufA as [row][blk(15)][p_pad(64)] f16, zero-padded
  // (covers jnp.pad 784->795 and K pad to 64).  Three lane-uniform regions
  // so the wave never diverges on the common path.
  // (1) p in [0,48): 8-wide runs, packed b128 LDS stores.
  for (int i = lane; i < 16 * 15 * 6; i += 32) {
    const int j   = i % 6;
    const int rb  = i / 6;            // row*15 + blk
    const int blk = rb % 15;
    const int p0  = j * 8;
    const int c0  = blk * 53 + p0;
    const int g   = grow0 + rb / 15;
    v8h pack;
    if (c0 + 7 < 784 && g < nB) {     // fast: only blk==14,j==5 falls through
      const float* src = x + (size_t)g * 784 + c0;
#pragma unroll
      for (int e = 0; e < 8; ++e) pack[e] = (_Float16)src[e];
    } else {
#pragma unroll
      for (int e = 0; e < 8; ++e) {
        float vv = 0.f;
        if (c0 + e < 784 && g < nB) vv = x[(size_t)g * 784 + c0 + e];
        pack[e] = (_Float16)vv;
      }
    }
    *(v8h*)(bufA + rb * 64 + p0) = pack;
  }
  // (2) p in [48,56): 5 data + 3 zero pads, one b128 store per run.
  for (int i = lane; i < 16 * 15; i += 32) {
    const int blk = i % 15;
    const int g   = grow0 + i / 15;
    const int c0  = blk * 53 + 48;
    v8h pack = {};
#pragma unroll
    for (int e = 0; e < 5; ++e) {
      float vv = 0.f;
      if (c0 + e < 784 && g < nB) vv = x[(size_t)g * 784 + c0 + e];
      pack[e] = (_Float16)vv;
    }
    *(v8h*)(bufA + i * 64 + 48) = pack;
  }
  // (3) p in [56,64): pure K-padding, zeros.
  {
    v8h z = {};
    for (int i = lane; i < 16 * 15; i += 32)
      *(v8h*)(bufA + i * 64 + 56) = z;
  }
  __syncthreads();

  //           SBLK KS QD NT | TSTR SD NT | FB1 FB2 | OUTF SBo PDo LAST
  run_layer<64, 2, 27, 2,   32, 27, 2,    0,  60,   392, 32, 27, false>(lane, bufA, bufT1, logits, wsh, b1);
  run_layer<32, 1,  9, 1,   16,  9, 1,   90, 105,   128, 16,  9, false>(lane, bufA, bufT1, logits, wsh, b2);
  run_layer<16, 1,  5, 1,    8,  5, 1,  120, 135,    64,  8,  5, false>(lane, bufA, bufT1, logits, wsh, b3);
  run_layer< 8, 1,  1, 1,    8,  1, 1,  150, 165,    10,  8,  5, true >(lane, bufA, bufT1, logits, wsh, b4);

  // log_softmax over 10 classes, one row per lane (lanes 0-15).
  if (lane < 16) {
    const int g = grow0 + lane;
    if (g < nB) {
      float m = -3.4e38f;
#pragma unroll
      for (int c = 0; c < 10; ++c) m = fmaxf(m, logits[lane * 16 + c]);
      float s = 0.f;
#pragma unroll
      for (int c = 0; c < 10; ++c) s += __expf(logits[lane * 16 + c] - m);
      const float lse = m + __logf(s);
#pragma unroll
      for (int c = 0; c < 10; ++c)
        out[(size_t)g * 10 + c] = logits[lane * 16 + c] - lse;
    }
  }
}

// ---------------------------------------------------------------------------
// Host launch: 8 weight-repack launches into d_ws, then the fused kernel.
// ws usage: 180 fragments * 1024B = 184320 bytes.
// ---------------------------------------------------------------------------
extern "C" void kernel_launch(void* const* d_in, const int* in_sizes, int n_in,
                              void* d_out, int out_size, void* d_ws, size_t ws_size,
                              hipStream_t stream) {
  (void)n_in; (void)out_size; (void)ws_size;
  _Float16* wsh = (_Float16*)d_ws;

  auto prep = [&](const void* w, int fragBase, int nblk, int ND, int KD,
                  int NT, int KS) {
    int total = nblk * NT * KS * 512;
    monarch_prep_frags<<<(total + 255) / 256, 256, 0, stream>>>(
        (const float*)w, wsh + (size_t)fragBase * 512, nblk, ND, KD, NT, KS);
  };

  // fragment bases: L1m1=0(60) L1m2=60(30) L2m1=90 L2m2=105 L3m1=120
  //                 L3m2=135 L4m1=150 L4m2=165  (total 180 frags)
  prep(d_in[1],   0, 15, 27, 53, 2, 2);  // fc1_w1 (k,q,p)
  prep(d_in[2],  60, 15, 27, 27, 2, 1);  // fc1_w2 (l,s,r)
  prep(d_in[4],  90, 15,  9, 27, 1, 1);  // fc2_w1
  prep(d_in[5], 105, 15,  9,  9, 1, 1);  // fc2_w2
  prep(d_in[7], 120, 15,  5,  9, 1, 1);  // fc3_w1
  prep(d_in[8], 135, 15,  5,  5, 1, 1);  // fc3_w2
  prep(d_in[10],150, 15,  1,  5, 1, 1);  // fc4_w1
  prep(d_in[11],165, 15,  1,  1, 1, 1);  // fc4_w2

  const int nB    = in_sizes[0] / 784;
  const int tiles = (nB + 15) / 16;
  monarch_mlp_main<<<tiles, 32, 0, stream>>>(
      (const float*)d_in[0], wsh,
      (const float*)d_in[3], (const float*)d_in[6],
      (const float*)d_in[9], (const float*)d_in[12],
      (float*)d_out, nB);
}
